// MultiTokenAttention_88656714924240
// MI455X (gfx1250) — matrix operand order repaired and checked by reading, not compiled, not run on hardware
//
#include <hip/hip_runtime.h>
#include <hip/hip_bf16.h>

// MultiTokenAttention for gfx1250 (MI455X), wave32 + WMMA bf16.
// B=2, S=1024, E=1024, H=16, HD=64, G=8 groups of GS=2 heads, 3x3 conv.

typedef __attribute__((ext_vector_type(16))) __bf16 v16bf;
typedef __attribute__((ext_vector_type(8)))  float  v8f;

#define SDIM 1024
#define EDIM 1024
#define NB   2
#define NH   16
#define HDD  64

// ---- WMMA fragment helpers (A and B share contiguous k order 16*laneHi + i) ----

// A fragment: 16x32 bf16 tile, row-major LDS [m][k], stride ld.
// lane m = lane&15; K = 16*laneHi + i.
__device__ inline v16bf frag_a(const __bf16* p, int ld) {
  int lane = threadIdx.x & 31;
  int m = lane & 15, hi = lane >> 4;
  v16bf a;
#pragma unroll
  for (int i = 0; i < 16; ++i) {
    int k = 16 * hi + i;
    a[i] = p[m * ld + k];
  }
  return a;
}

// B fragment: 32x16 bf16 tile, row-major LDS [k][n], stride ld.
// lane: n = lane&15; K = 16*laneHi + i.
__device__ inline v16bf frag_b(const __bf16* p, int ld) {
  int lane = threadIdx.x & 31;
  int n = lane & 15, hi = lane >> 4;
  v16bf b;
#pragma unroll
  for (int i = 0; i < 16; ++i) b[i] = p[(hi * 16 + i) * ld + n];
  return b;
}

// B fragment gathered from a row-major [n][k] tile (i.e. B = tile^T).
__device__ inline v16bf frag_bt(const __bf16* p, int ld) {
  int lane = threadIdx.x & 31;
  int n = lane & 15, hi = lane >> 4;
  v16bf b;
#pragma unroll
  for (int i = 0; i < 16; ++i) b[i] = p[n * ld + hi * 16 + i];
  return b;
}

__device__ inline v8f wmma_bf16(v16bf a, v16bf b, v8f c) {
  return __builtin_amdgcn_wmma_f32_16x16x32_bf16(false, a, false, b, (short)0, c,
                                                 false, false);
}

// ---- Kernel 1: fused QKV projections, X@W+b -> bf16 head-major (b,h,s,d) ----
__global__ __launch_bounds__(256) void k_proj(
    const float* Q, const float* K, const float* V,
    const float* Wq, const float* Wk, const float* Wv,
    const float* bq, const float* bk, const float* bv,
    __bf16* qo, __bf16* ko, __bf16* vo) {
  const float *X, *W, *bias;
  __bf16* out;
  if (blockIdx.z == 0) { X = Q; W = Wq; bias = bq; out = qo; }
  else if (blockIdx.z == 1) { X = K; W = Wk; bias = bk; out = ko; }
  else { X = V; W = Wv; bias = bv; out = vo; }

  __shared__ __bf16 As[128 * 34];
  __shared__ __bf16 Bs[32 * 130];
  int t = threadIdx.x;
  int M0 = blockIdx.y * 128, N0 = blockIdx.x * 128;
  int w = t >> 5, wm = w >> 1, wn = w & 1;

  v8f acc[8];
#pragma unroll
  for (int i = 0; i < 8; ++i) acc[i] = {};

  for (int kb = 0; kb < EDIM; kb += 32) {
#pragma unroll
    for (int it = 0; it < 16; ++it) {
      int idx = t + it * 256;
      int r = idx >> 5, c = idx & 31;
      As[r * 34 + c] = (__bf16)X[(M0 + r) * EDIM + kb + c];
      int rb = idx >> 7, cb = idx & 127;
      Bs[rb * 130 + cb] = (__bf16)W[(kb + rb) * EDIM + N0 + cb];
    }
    __syncthreads();
    v16bf a0 = frag_a(As + (wm * 32) * 34, 34);
    v16bf a1 = frag_a(As + (wm * 32 + 16) * 34, 34);
#pragma unroll
    for (int j = 0; j < 4; ++j) {
      v16bf bj = frag_b(Bs + wn * 64 + j * 16, 130);
      acc[j]     = wmma_bf16(a0, bj, acc[j]);
      acc[4 + j] = wmma_bf16(a1, bj, acc[4 + j]);
    }
    __syncthreads();
  }

  int lane = t & 31, n = lane & 15, hi = lane >> 4;
#pragma unroll
  for (int i = 0; i < 2; ++i)
#pragma unroll
    for (int j = 0; j < 4; ++j)
#pragma unroll
      for (int r = 0; r < 8; ++r) {
        int m  = M0 + wm * 32 + i * 16 + r + 8 * hi;
        int nc = N0 + wn * 64 + j * 16 + n;
        float val = acc[i * 4 + j][r] + bias[nc];
        int bb = m >> 10, s = m & 1023;
        int h = nc >> 6, d = nc & 63;
        out[(((bb * NH + h) * SDIM) + s) * HDD + d] = (__bf16)val;
      }
}

// ---- Kernel 2: per-head logits = scale * (q k^T), causal-zeroed, bf16 ----
__global__ __launch_bounds__(256) void k_logits(const __bf16* qh,
                                                const __bf16* kh,
                                                __bf16* logits) {
  int bx = blockIdx.x, by = blockIdx.y, z = blockIdx.z;  // z = b*NH + h
  if (bx > by) return;  // strictly upper tiles are never read

  __shared__ __bf16 Qs[128 * 66];
  __shared__ __bf16 Ks[128 * 66];
  int t = threadIdx.x;
#pragma unroll
  for (int it = 0; it < 32; ++it) {
    int idx = t + it * 256;
    int r = idx >> 6, c = idx & 63;
    Qs[r * 66 + c] = qh[((z * SDIM) + by * 128 + r) * HDD + c];
    Ks[r * 66 + c] = kh[((z * SDIM) + bx * 128 + r) * HDD + c];
  }
  __syncthreads();

  int w = t >> 5, wm = w >> 1, wn = w & 1;
  v8f acc[8];
#pragma unroll
  for (int i = 0; i < 8; ++i) acc[i] = {};

#pragma unroll
  for (int ks = 0; ks < 2; ++ks) {
    v16bf a0 = frag_a(Qs + (wm * 32) * 66 + ks * 32, 66);
    v16bf a1 = frag_a(Qs + (wm * 32 + 16) * 66 + ks * 32, 66);
#pragma unroll
    for (int j = 0; j < 4; ++j) {
      v16bf bj = frag_bt(Ks + (wn * 64 + j * 16) * 66 + ks * 32, 66);
      acc[j]     = wmma_bf16(a0, bj, acc[j]);
      acc[4 + j] = wmma_bf16(a1, bj, acc[4 + j]);
    }
  }

  int lane = t & 31, n = lane & 15, hi = lane >> 4;
#pragma unroll
  for (int i = 0; i < 2; ++i)
#pragma unroll
    for (int j = 0; j < 4; ++j)
#pragma unroll
      for (int r = 0; r < 8; ++r) {
        int q  = by * 128 + wm * 32 + i * 16 + r + 8 * hi;
        int kc = bx * 128 + wn * 64 + j * 16 + n;
        float val = acc[i * 4 + j][r] * 0.125f;
        if (kc > q) val = 0.f;
        logits[(z * SDIM + q) * SDIM + kc] = (__bf16)val;
      }
}

// ---- Kernel 3: fused conv3x3 + bias + softmax + head-mix + attn@V + DyT ----
// grid: (S/16, G=8, B). block: 256 (8 waves).
__global__ __launch_bounds__(256) void k_attn(
    const __bf16* logits, const __bf16* vh, const float* kq_w,
    const float* kq_b, const float* mix_w, const float* mix_b,
    const float* dyt_alpha, const float* dyt_w, const float* dyt_b,
    const float* depth_scale, __bf16* outh) {
  int qt = blockIdx.x, g = blockIdx.y, b = blockIdx.z;
  int q0 = qt * 16;
  int h0 = g * 2;

  __shared__ __bf16 Lc[2][18][36];     // logits halo chunk, both heads
  __shared__ float  Pbuf[2][16][32];   // softmax probs for the chunk
  __shared__ __bf16 Am[2][16][32];     // mixed attn A-tiles (per output head)
  __shared__ __bf16 Vt[2][32][68];     // V chunk (per output head)
  __shared__ float  redm[32][8], reds[32][8];
  __shared__ float  rowM[32], rowS[32];

  int t = threadIdx.x;
  int head = t >> 7;       // which input head of the group this thread convs
  int tid = t & 127;
  int jrow = tid >> 3;     // q-row in tile, 0..15
  int kk0 = (tid & 7) * 4; // 4 consecutive cols per thread per chunk
  int h = h0 + head;

  float w9[9];
#pragma unroll
  for (int i = 0; i < 9; ++i) w9[i] = kq_w[h * 9 + i];
  float kqb = kq_b[h];

  // ---------------- pass 1: online max/sum of exp(conv) ----------------
  float mrun = -3.0e38f, srun = 0.f;
  for (int ch = 0; ch < 32; ++ch) {
    int c0 = ch * 32;
    for (int idx = t; idx < 2 * 18 * 34; idx += 256) {
      int hh = idx / (18 * 34);
      int rem = idx - hh * (18 * 34);
      int j = rem / 34, cc = rem - j * 34;
      int r = q0 - 2 + j;
      int c = c0 - 1 + cc;
      float v = 0.f;
      if (r >= 0 && c >= 0 && c < SDIM && c <= r)
        v = (float)logits[((size_t)((b * NH + h0 + hh) * SDIM) + r) * SDIM + c];
      Lc[hh][j][cc] = (__bf16)v;
    }
    __syncthreads();
#pragma unroll
    for (int u = 0; u < 4; ++u) {
      int kk = kk0 + u;
      float cv = kqb;
#pragma unroll
      for (int di = 0; di < 3; ++di)
#pragma unroll
        for (int dj = 0; dj < 3; ++dj)
          cv += w9[di * 3 + dj] * (float)Lc[head][jrow + di][kk + dj];
      float mnew = fmaxf(mrun, cv);
      srun = srun * __expf(mrun - mnew) + __expf(cv - mnew);
      mrun = mnew;
    }
    __syncthreads();
  }
  redm[head * 16 + jrow][tid & 7] = mrun;
  reds[head * 16 + jrow][tid & 7] = srun;
  __syncthreads();
  if (t < 32) {
    float mm = -3.0e38f;
#pragma unroll
    for (int i = 0; i < 8; ++i) mm = fmaxf(mm, redm[t][i]);
    float ss = 0.f;
#pragma unroll
    for (int i = 0; i < 8; ++i) ss += reds[t][i] * __expf(redm[t][i] - mm);
    rowM[t] = mm;
    rowS[t] = ss;
  }
  __syncthreads();
  float rM = rowM[head * 16 + jrow];
  float invS = 1.0f / rowS[head * 16 + jrow];

  // ---------------- pass 2: probs -> mix -> WMMA with V ----------------
  v8f acc = {};
  int w = t >> 5;
  int o = w >> 2, nf = w & 3;  // output head within group, 16-col fragment
  for (int ch = 0; ch < 32; ++ch) {
    int c0 = ch * 32;
    for (int idx = t; idx < 2 * 18 * 34; idx += 256) {
      int hh = idx / (18 * 34);
      int rem = idx - hh * (18 * 34);
      int j = rem / 34, cc = rem - j * 34;
      int r = q0 - 2 + j;
      int c = c0 - 1 + cc;
      float v = 0.f;
      if (r >= 0 && c >= 0 && c < SDIM && c <= r)
        v = (float)logits[((size_t)((b * NH + h0 + hh) * SDIM) + r) * SDIM + c];
      Lc[hh][j][cc] = (__bf16)v;
    }
    for (int idx = t; idx < 2 * 32 * 64; idx += 256) {
      int hh = idx >> 11;
      int kk = (idx >> 6) & 31;
      int d = idx & 63;
      Vt[hh][kk][d] =
          vh[((size_t)((b * NH + h0 + hh) * SDIM) + c0 + kk) * HDD + d];
    }
    __syncthreads();
#pragma unroll
    for (int u = 0; u < 4; ++u) {
      int kk = kk0 + u;
      float cv = kqb;
#pragma unroll
      for (int di = 0; di < 3; ++di)
#pragma unroll
        for (int dj = 0; dj < 3; ++dj)
          cv += w9[di * 3 + dj] * (float)Lc[head][jrow + di][kk + dj];
      Pbuf[head][jrow][kk] = __expf(cv - rM) * invS;
    }
    __syncthreads();
    for (int idx = t; idx < 1024; idx += 256) {
      int oo = idx >> 9;
      int rr = (idx >> 5) & 15;
      int kk = idx & 31;
      float p0 = Pbuf[0][rr][kk], p1 = Pbuf[1][rr][kk];
      float mixed = mix_w[(g * 2 + oo) * 2 + 0] * p0 +
                    mix_w[(g * 2 + oo) * 2 + 1] * p1 + mix_b[h0 + oo];
      Am[oo][rr][kk] = (__bf16)mixed;
    }
    __syncthreads();
    v16bf a = frag_a(&Am[o][0][0], 32);
    v16bf bb = frag_b(&Vt[o][0][nf * 16], 68);
    acc = wmma_bf16(a, bb, acc);
    __syncthreads();
  }

  // DyT epilogue, store to (B,S,E) bf16
  float alpha = dyt_alpha[0], dscale = depth_scale[0];
  int lane = t & 31, n = lane & 15, hi = lane >> 4;
  int d = nf * 16 + n;
  float dw = dyt_w[d], db = dyt_b[d];
  int ho = h0 + o;
#pragma unroll
  for (int r = 0; r < 8; ++r) {
    int q = q0 + r + 8 * hi;
    float val = acc[r];
    val = tanhf(alpha * val) * dw + db;
    val *= dscale;
    outh[(size_t)(b * SDIM + q) * EDIM + ho * HDD + d] = (__bf16)val;
  }
}

// ---- Kernel 4: final projection, outh(bf16) @ Wo + bo -> fp32 ----
__global__ __launch_bounds__(256) void k_out(const __bf16* A, const float* Wo,
                                             const float* bo, float* out) {
  __shared__ __bf16 As[128 * 34];
  __shared__ __bf16 Bs[32 * 130];
  int t = threadIdx.x;
  int M0 = blockIdx.y * 128, N0 = blockIdx.x * 128;
  int w = t >> 5, wm = w >> 1, wn = w & 1;

  v8f acc[8];
#pragma unroll
  for (int i = 0; i < 8; ++i) acc[i] = {};

  for (int kb = 0; kb < EDIM; kb += 32) {
#pragma unroll
    for (int it = 0; it < 16; ++it) {
      int idx = t + it * 256;
      int r = idx >> 5, c = idx & 31;
      As[r * 34 + c] = A[(size_t)(M0 + r) * EDIM + kb + c];
      int rb = idx >> 7, cb = idx & 127;
      Bs[rb * 130 + cb] = (__bf16)Wo[(kb + rb) * EDIM + N0 + cb];
    }
    __syncthreads();
    v16bf a0 = frag_a(As + (wm * 32) * 34, 34);
    v16bf a1 = frag_a(As + (wm * 32 + 16) * 34, 34);
#pragma unroll
    for (int j = 0; j < 4; ++j) {
      v16bf bj = frag_b(Bs + wn * 64 + j * 16, 130);
      acc[j]     = wmma_bf16(a0, bj, acc[j]);
      acc[4 + j] = wmma_bf16(a1, bj, acc[4 + j]);
    }
    __syncthreads();
  }

  int lane = t & 31, n = lane & 15, hi = lane >> 4;
#pragma unroll
  for (int i = 0; i < 2; ++i)
#pragma unroll
    for (int j = 0; j < 4; ++j)
#pragma unroll
      for (int r = 0; r < 8; ++r) {
        int m  = M0 + wm * 32 + i * 16 + r + 8 * hi;
        int nc = N0 + wn * 64 + j * 16 + n;
        out[(size_t)m * EDIM + nc] = acc[i * 4 + j][r] + bo[nc];
      }
}

extern "C" void kernel_launch(void* const* d_in, const int* in_sizes, int n_in,
                              void* d_out, int out_size, void* d_ws,
                              size_t ws_size, hipStream_t stream) {
  const float* Q = (const float*)d_in[0];
  const float* K = (const float*)d_in[1];
  const float* V = (const float*)d_in[2];
  const float* Wq = (const float*)d_in[3];
  const float* bq = (const float*)d_in[4];
  const float* Wk = (const float*)d_in[5];
  const float* bk = (const float*)d_in[6];
  const float* Wv = (const float*)d_in[7];
  const float* bv = (const float*)d_in[8];
  const float* Wo = (const float*)d_in[9];
  const float* bo = (const float*)d_in[10];
  const float* kq_w = (const float*)d_in[11];
  const float* kq_b = (const float*)d_in[12];
  const float* mix_w = (const float*)d_in[13];
  const float* mix_b = (const float*)d_in[14];
  const float* dyt_alpha = (const float*)d_in[15];
  const float* dyt_w = (const float*)d_in[16];
  const float* dyt_b = (const float*)d_in[17];
  const float* depth_scale = (const float*)d_in[18];
  float* out = (float*)d_out;

  // workspace partition (bf16): q,k,v head-major; logits; outh
  __bf16* qh = (__bf16*)d_ws;
  __bf16* kh = qh + (size_t)NB * NH * SDIM * HDD;
  __bf16* vh = kh + (size_t)NB * NH * SDIM * HDD;
  __bf16* logits = vh + (size_t)NB * NH * SDIM * HDD;
  __bf16* outh = logits + (size_t)NB * NH * SDIM * SDIM;

  k_proj<<<dim3(EDIM / 128, (NB * SDIM) / 128, 3), 256, 0, stream>>>(
      Q, K, V, Wq, Wk, Wv, bq, bk, bv, qh, kh, vh);
  k_logits<<<dim3(SDIM / 128, SDIM / 128, NB * NH), 256, 0, stream>>>(qh, kh,
                                                                      logits);
  k_attn<<<dim3(SDIM / 16, NH / 2, NB), 256, 0, stream>>>(
      logits, vh, kq_w, kq_b, mix_w, mix_b, dyt_alpha, dyt_w, dyt_b,
      depth_scale, outh);
  k_out<<<dim3(EDIM / 128, (NB * SDIM) / 128, 1), 256, 0, stream>>>(outh, Wo,
                                                                    bo, out);
}
